// NLPBaseNet_23338852286479
// MI455X (gfx1250) — matrix operand order repaired and checked by reading, not compiled, not run on hardware
//
#include <hip/hip_runtime.h>
#include <hip/hip_bf16.h>
#include <math.h>

// ---------------------------------------------------------------------------
// Problem dimensions (fixed by the reference)
// ---------------------------------------------------------------------------
#define Bsz 64
#define Tsz 1024
#define Isz 256
#define Hsz 512
#define Gsz (3 * Hsz)      // 1536
#define Osz 256
#define MTOT (Bsz * Tsz)   // 65536

#define NBLK 32            // blocks in persistent GRU layer kernel (16 cols each)
#define NCOL 16            // H-columns per block

typedef __bf16 bf16t;
typedef __attribute__((ext_vector_type(16))) __bf16 v16bf;
typedef __attribute__((ext_vector_type(8)))  __bf16 v8bf;
typedef __attribute__((ext_vector_type(8)))  float  v8f;

// ---------------------------------------------------------------------------
// WMMA helpers (CDNA5 16x16x32 bf16, fp32 accumulate)
// ---------------------------------------------------------------------------
__device__ __forceinline__ v8f wmma_bf16(v16bf a, v16bf b, v8f c) {
  // 8 args: (neg_a, A, neg_b, B, c_mod, C, reuse_a, reuse_b)
  return __builtin_amdgcn_wmma_f32_16x16x32_bf16(false, a, false, b,
                                                 (short)0, c, false, false);
}

// A-matrix fragment, 16x32 bf16, row-major source [*, K].
// lane l holds row (l&15); 16 consecutive K values at k0 + 16*(l>=16)
__device__ __forceinline__ v16bf load_a_frag(const bf16t* __restrict__ row,
                                             int k0, int hi) {
  v8bf lo  = *(const v8bf*)(row + k0 + hi * 16);
  v8bf hi8 = *(const v8bf*)(row + k0 + hi * 16 + 8);
  v16bf a;
#pragma unroll
  for (int i = 0; i < 8; ++i) { a[i] = lo[i]; a[i + 8] = hi8[i]; }
  return a;
}

// B-matrix fragment, 32x16 bf16; source is W row-major [N,K] (i.e. B^T).
// lane l holds column (l&15); 16 consecutive K values at k0 + 16*(l>=16).
__device__ __forceinline__ v16bf load_b_frag(const bf16t* __restrict__ W,
                                             int wrow, int K, int k0, int hi) {
  return *(const v16bf*)(W + (size_t)wrow * K + k0 + hi * 16);
}

// Same fragment layout, but sourced from an LDS-resident weight slice
// [48 rows x 512] bf16 -> two ds_load_b128 per lane.
__device__ __forceinline__ v16bf load_b_lds(const bf16t* wlds, int lrow,
                                            int k0, int hi) {
  return *(const v16bf*)(wlds + lrow * Hsz + k0 + hi * 16);
}

__device__ __forceinline__ float sigmoidf_fast(float x) {
  return 1.0f / (1.0f + __expf(-x));
}

// ---------------------------------------------------------------------------
// Elementwise kernels
// ---------------------------------------------------------------------------
__global__ void k_f32_to_bf16(const float* __restrict__ s,
                              bf16t* __restrict__ d, size_t n) {
  size_t i = (size_t)blockIdx.x * blockDim.x + threadIdx.x;
  size_t stride = (size_t)gridDim.x * blockDim.x;
  for (; i < n; i += stride) d[i] = (bf16t)s[i];
}

// Zero the t=0 hidden-state ping buffer and reset the step barrier.
__global__ void k_init_layer(bf16t* __restrict__ hb, unsigned* __restrict__ bar,
                             int n) {
  int i = blockIdx.x * blockDim.x + threadIdx.x;
  if (i == 0) *bar = 0u;
  if (i < n) hb[i] = (bf16t)0.0f;
}

// ---------------------------------------------------------------------------
// Generic bf16 GEMM + bias:   C[M,N](fp32) = A[M,K](bf16) * W[N,K]^T + bias[N]
// Block = 256 threads = 8 waves; each wave computes a 32(M) x 64(N) tile.
// Grid: (N/64, M/256). M%256==0, N%64==0, K%32==0.
// ---------------------------------------------------------------------------
__global__ __launch_bounds__(256)
void k_gemm_bf16(const bf16t* __restrict__ A, const bf16t* __restrict__ W,
                 const float* __restrict__ bias, float* __restrict__ C,
                 int M, int N, int K) {
  const int lane = threadIdx.x & 31;
  const int wave = threadIdx.x >> 5;
  const int lh = lane & 15;
  const int hi = lane >> 4;

  const int n0 = blockIdx.x * 64;
  const int m0 = (blockIdx.y * 8 + wave) * 32;

  const bf16t* arow0 = A + (size_t)(m0 + lh) * K;
  const bf16t* arow1 = A + (size_t)(m0 + 16 + lh) * K;

  v8f acc[2][4] = {};

  for (int k0 = 0; k0 < K; k0 += 32) {
    v16bf a0 = load_a_frag(arow0, k0, hi);
    v16bf a1 = load_a_frag(arow1, k0, hi);
#pragma unroll
    for (int j = 0; j < 4; ++j) {
      v16bf b = load_b_frag(W, n0 + 16 * j + lh, K, k0, hi);
      acc[0][j] = wmma_bf16(a0, b, acc[0][j]);
      acc[1][j] = wmma_bf16(a1, b, acc[1][j]);
    }
  }

  // C/D fragment layout: lane l, vgpr v  ->  m = v + 8*hi, n = l&15
#pragma unroll
  for (int mt = 0; mt < 2; ++mt) {
#pragma unroll
    for (int j = 0; j < 4; ++j) {
      const int ncol = n0 + 16 * j + lh;
      const float bv = bias[ncol];
#pragma unroll
      for (int v = 0; v < 8; ++v) {
        const int m = m0 + mt * 16 + v + 8 * hi;
        C[(size_t)m * N + ncol] = acc[mt][j][v] + bv;
      }
    }
  }
}

// ---------------------------------------------------------------------------
// Agent-scope step barrier for the persistent layer kernel (32 co-resident
// blocks; release-add / acquire-spin with s_sleep).
// ---------------------------------------------------------------------------
__device__ __forceinline__ void step_barrier(unsigned* bar, unsigned target) {
  __threadfence();
  __syncthreads();
  if (threadIdx.x == 0) {
    __hip_atomic_fetch_add(bar, 1u, __ATOMIC_RELEASE, __HIP_MEMORY_SCOPE_AGENT);
    while (__hip_atomic_load(bar, __ATOMIC_ACQUIRE,
                             __HIP_MEMORY_SCOPE_AGENT) < target)
      __builtin_amdgcn_s_sleep(1);
  }
  __syncthreads();
  __threadfence();
}

// ---------------------------------------------------------------------------
// Persistent GRU layer kernel: runs the ENTIRE T=1024 recurrence in one
// launch. 32 blocks x 128 threads (4 waves). Block b owns H-columns
// [16b, 16b+16) for all three gates:
//   - its W_hh slice (3x16x512 bf16 = 48 KB) is staged into LDS once and is
//     read via ds_load_b128 fragments for all 1024 steps (CDNA5 320KB LDS),
//   - fp32 hidden state lives in registers (each lane owns 8 (batch,col)
//     elements), only the bf16 ping-pong copy goes through global memory
//     for the cross-WGP A-fragment reads,
//   - blocks synchronize once per timestep with an atomic barrier instead
//     of 1024 kernel launches.
// Wave w handles batch rows [16w, 16w+16): 3 gate accumulators per wave,
// 16 k-iterations x 3 WMMAs per step.
// ---------------------------------------------------------------------------
__global__ __launch_bounds__(128)
void k_gru_layer(const bf16t* __restrict__ hb0,   // [B,H] bf16 ping (t even)
                 bf16t* __restrict__ hb1,         // [B,H] bf16 pong (t odd)
                 const bf16t* __restrict__ W_hh,  // [3H,H] bf16
                 const float* __restrict__ b_hh,  // [3H]
                 const float* __restrict__ xg,    // [B*T,3H] fp32, row b*T+t
                 bf16t* __restrict__ out_bf,      // [B,T,H] bf16
                 float* __restrict__ hT_out,      // [B,H] fp32 (final state)
                 unsigned* __restrict__ bar) {
  __shared__ __attribute__((aligned(32))) bf16t wlds[3 * NCOL * Hsz]; // 48 KB

  const int lane = threadIdx.x & 31;
  const int wave = threadIdx.x >> 5;   // 0..3 -> batch rows 16*wave..+15
  const int lh = lane & 15;
  const int hi = lane >> 4;
  const int c0 = blockIdx.x * NCOL;    // column chunk within H
  const int col = c0 + lh;             // this lane's H-column

  // ---- stage this block's W_hh slice into LDS (rows: g*16 + local row) ----
  // 48 rows x 512 bf16 = 3072 16-byte chunks, 24 per thread.
  for (int ci = threadIdx.x; ci < 3 * NCOL * (Hsz / 8); ci += blockDim.x) {
    const int lrow = ci >> 6;          // 0..47   (Hsz/8 = 64 chunks per row)
    const int ck = ci & 63;
    const int g = lrow >> 4;
    const int rl = lrow & 15;
    *(v8bf*)(wlds + lrow * Hsz + ck * 8) =
        *(const v8bf*)(W_hh + (size_t)(g * Hsz + c0 + rl) * Hsz + ck * 8);
  }
  __syncthreads();

  // Per-lane loop-invariant gate biases.
  const float br = b_hh[col];
  const float bz = b_hh[Hsz + col];
  const float bn = b_hh[2 * Hsz + col];

  // fp32 hidden state, register-resident: hreg[v] = h[16*wave + v + 8*hi, col]
  float hreg[8];
#pragma unroll
  for (int v = 0; v < 8; ++v) hreg[v] = 0.0f;

  const bf16t* arow_base0 = hb0 + (size_t)(wave * 16 + lh) * Hsz;
  const bf16t* arow_base1 = hb1 + (size_t)(wave * 16 + lh) * Hsz;

  for (int t = 0; t < Tsz; ++t) {
    const bf16t* arow = (t & 1) ? arow_base1 : arow_base0;
    bf16t* hnext = (t & 1) ? (bf16t*)hb0 : hb1;

    // Prefetch this step's xg rows so the epilogue loads hit cache.
#pragma unroll
    for (int v = 0; v < 8; ++v) {
      const size_t xrow = ((size_t)(wave * 16 + v + 8 * hi) * Tsz + t) * Gsz;
      __builtin_prefetch(xg + xrow + col, 0, 1);
      __builtin_prefetch(xg + xrow + Hsz + col, 0, 1);
      __builtin_prefetch(xg + xrow + 2 * Hsz + col, 0, 1);
    }

    // ---- hg = h_t @ W_hh^T for this block's 16 columns, 3 gates ----
    v8f acc[3] = {};
#pragma unroll 4
    for (int k0 = 0; k0 < Hsz; k0 += 32) {
      v16bf a = load_a_frag(arow, k0, hi);
#pragma unroll
      for (int g = 0; g < 3; ++g) {
        v16bf b = load_b_lds(wlds, g * NCOL + lh, k0, hi);
        acc[g] = wmma_bf16(a, b, acc[g]);
      }
    }

    // ---- gate epilogue on the C/D fragment layout ----
#pragma unroll
    for (int v = 0; v < 8; ++v) {
      const int brow = wave * 16 + v + 8 * hi;          // batch index
      const size_t xrow = ((size_t)brow * Tsz + t) * Gsz;
      const float hr = acc[0][v] + br;
      const float hz = acc[1][v] + bz;
      const float hn = acc[2][v] + bn;
      const float r = sigmoidf_fast(xg[xrow + col] + hr);
      const float z = sigmoidf_fast(xg[xrow + Hsz + col] + hz);
      const float nn = tanhf(xg[xrow + 2 * Hsz + col] + r * hn);
      const float hnew = (1.0f - z) * nn + z * hreg[v];
      hreg[v] = hnew;
      hnext[(size_t)brow * Hsz + col] = (bf16t)hnew;
      out_bf[((size_t)brow * Tsz + t) * Hsz + col] = (bf16t)hnew;
      if (t == Tsz - 1) hT_out[(size_t)brow * Hsz + col] = hnew;
    }

    // One barrier per step: h(t+1) writes visible before any t+1 reads.
    step_barrier(bar, (unsigned)NBLK * (unsigned)(t + 1));
  }
}

// ---------------------------------------------------------------------------
// Host-side orchestration
// ---------------------------------------------------------------------------
extern "C" void kernel_launch(void* const* d_in, const int* in_sizes, int n_in,
                              void* d_out, int out_size, void* d_ws,
                              size_t ws_size, hipStream_t stream) {
  (void)in_sizes; (void)n_in; (void)out_size; (void)ws_size;

  const float* x     = (const float*)d_in[0];
  const float* W_ih0 = (const float*)d_in[1];
  const float* W_hh0 = (const float*)d_in[2];
  const float* b_ih0 = (const float*)d_in[3];
  const float* b_hh0 = (const float*)d_in[4];
  const float* W_ih1 = (const float*)d_in[5];
  const float* W_hh1 = (const float*)d_in[6];
  const float* b_ih1 = (const float*)d_in[7];
  const float* b_hh1 = (const float*)d_in[8];
  const float* fc_w  = (const float*)d_in[9];
  const float* fc_b  = (const float*)d_in[10];

  float* out_fc = (float*)d_out;                        // [B*T, O]
  float* hidden = out_fc + (size_t)MTOT * Osz;          // [2, B, H]

  // Workspace carving (256B-aligned slabs)
  char* ws = (char*)d_ws;
  size_t off = 0;
  auto carve = [&](size_t bytes) -> void* {
    void* p = ws + off;
    off = (off + bytes + 255) & ~(size_t)255;
    return p;
  };
  bf16t* x_bf    = (bf16t*)carve((size_t)MTOT * Isz * 2);
  bf16t* wih0_bf = (bf16t*)carve((size_t)Gsz * Isz * 2);
  bf16t* whh0_bf = (bf16t*)carve((size_t)Gsz * Hsz * 2);
  bf16t* wih1_bf = (bf16t*)carve((size_t)Gsz * Hsz * 2);
  bf16t* whh1_bf = (bf16t*)carve((size_t)Gsz * Hsz * 2);
  bf16t* fcw_bf  = (bf16t*)carve((size_t)Osz * Hsz * 2);
  bf16t* out0_bf = (bf16t*)carve((size_t)MTOT * Hsz * 2);
  bf16t* out1_bf = (bf16t*)carve((size_t)MTOT * Hsz * 2);
  bf16t* hb0     = (bf16t*)carve((size_t)Bsz * Hsz * 2);
  bf16t* hb1     = (bf16t*)carve((size_t)Bsz * Hsz * 2);
  unsigned* bar  = (unsigned*)carve(256);
  float* xg      = (float*)carve((size_t)MTOT * Gsz * 4); // reused per layer

  // ---- fp32 -> bf16 conversions -------------------------------------------
  auto conv = [&](const float* s, bf16t* d, size_t n) {
    int blocks = (int)((n + 256 * 8 - 1) / (256 * 8));
    if (blocks > 4096) blocks = 4096;
    k_f32_to_bf16<<<blocks, 256, 0, stream>>>(s, d, n);
  };
  conv(x, x_bf, (size_t)MTOT * Isz);
  conv(W_ih0, wih0_bf, (size_t)Gsz * Isz);
  conv(W_hh0, whh0_bf, (size_t)Gsz * Hsz);
  conv(W_ih1, wih1_bf, (size_t)Gsz * Hsz);
  conv(W_hh1, whh1_bf, (size_t)Gsz * Hsz);
  conv(fc_w, fcw_bf, (size_t)Osz * Hsz);

  const int hn = Bsz * Hsz;

  // ==================== Layer 0 ====================
  {
    dim3 g(Gsz / 64, MTOT / 256);   // xg0 = x @ W_ih0^T + b_ih0  (K=256)
    k_gemm_bf16<<<g, 256, 0, stream>>>(x_bf, wih0_bf, b_ih0, xg,
                                       MTOT, Gsz, Isz);
  }
  k_init_layer<<<(hn + 255) / 256, 256, 0, stream>>>(hb0, bar, hn);
  k_gru_layer<<<NBLK, 128, 0, stream>>>(hb0, hb1, whh0_bf, b_hh0, xg,
                                        out0_bf, hidden, bar);

  // ==================== Layer 1 ====================
  {
    dim3 g(Gsz / 64, MTOT / 256);   // xg1 = out0 @ W_ih1^T + b_ih1  (K=512)
    k_gemm_bf16<<<g, 256, 0, stream>>>(out0_bf, wih1_bf, b_ih1, xg,
                                       MTOT, Gsz, Hsz);
  }
  k_init_layer<<<(hn + 255) / 256, 256, 0, stream>>>(hb0, bar, hn);
  k_gru_layer<<<NBLK, 128, 0, stream>>>(hb0, hb1, whh1_bf, b_hh1, xg,
                                        out1_bf, hidden + hn, bar);

  // ==================== FC head ====================
  {
    dim3 g(Osz / 64, MTOT / 256);   // out = out1 @ fc_w^T + fc_b  (K=512)
    k_gemm_bf16<<<g, 256, 0, stream>>>(out1_bf, fcw_bf, fc_b, out_fc,
                                       MTOT, Osz, Hsz);
  }
}
